// DiseaseGraphEncoder_10342281248978
// MI455X (gfx1250) — matrix operand order repaired and checked by reading, not compiled, not run on hardware
//
#include <hip/hip_runtime.h>
#include <hip/hip_bf16.h>

// ---------------------------------------------------------------------------
// DiseaseGraphEncoder on MI455X (gfx1250, wave32, WMMA).
//   h   = (features padded 15->32) @ W_in(bf16) + b_in                  [WMMA]
//   per layer L:
//     hb  = h @ Bconcat_L (64 x 256, bf16)                              [WMMA]
//     agg = scatter-add over edges of sum_b coeffs[etype,b]*hb[src,b,:] [atomics, L2]
//     h   = relu(h @ Wself_L(bf16) + bself_L + agg)                     [WMMA]
//   graph_emb = MLP(mean||max pooling)  (deterministic 2-stage reduce)
//   node_emb  = h @ Wnp(bf16) + bnp                                     [WMMA]
// B weights are pre-swizzled into WMMA fragment-major order: each lane's
// 16 bf16 fragment is 32 contiguous bytes -> 2x global_load_b128, no repack.
// ---------------------------------------------------------------------------

typedef __attribute__((ext_vector_type(16))) __bf16 v16bf;
typedef __attribute__((ext_vector_type(8)))  float  v8f;

struct bfrag32 { uint4 a, b; };   // 32 bytes = one lane's B fragment

__device__ __forceinline__ unsigned short bf_bits(float x) {
  unsigned u = __builtin_bit_cast(unsigned, x);
  u += 0x7FFFu + ((u >> 16) & 1u);               // round-to-nearest-even
  return (unsigned short)(u >> 16);
}
__device__ __forceinline__ __bf16 f2bf(float x) {
  unsigned short h = bf_bits(x);
  return __builtin_bit_cast(__bf16, h);
}
__device__ __forceinline__ unsigned pack2bf(float a, float b) {
  return (unsigned)bf_bits(a) | ((unsigned)bf_bits(b) << 16);
}

// ---------------------------------------------------------------------------
// Pack a KxN weight (f32) into fragment-major bf16:
//   out[ ((nt*(KT/32)+kc)*32 + lane)*16 + j ] = B[kc*32 + (lane<16?0:16) + j,
//                                                 nt*16 + (lane&15)]
// mode 0: plain row-major (Kin x NT), rows >= Kin zero-padded.
// mode 1: bases concat: B[k, col] = bases[col>>6][k][col&63]  (NT=256)
// ---------------------------------------------------------------------------
template <int KT, int NT>
__global__ void k_pack_B(const float* __restrict__ in, __bf16* __restrict__ out,
                         int Kin, int mode) {
  constexpr int KC = KT / 32;
  int o = blockIdx.x * blockDim.x + threadIdx.x;
  if (o >= KT * NT) return;
  int j   = o & 15;
  int l   = (o >> 4) & 31;
  int kc  = (o >> 9) % KC;
  int nt  = o / (512 * KC);
  int col = nt * 16 + (l & 15);
  int k   = kc * 32 + ((l < 16) ? 0 : 16) + j;
  float v = 0.0f;
  if (k < Kin) {
    if (mode == 0) v = in[(size_t)k * NT + col];
    else           v = in[(size_t)(col >> 6) * 4096 + (size_t)k * 64 + (col & 63)];
  }
  out[o] = f2bf(v);
}

// ---------------------------------------------------------------------------
// bf16 WMMA GEMM: C(MxN,f32) = A(MxK,f32) * Bswz(KxN,bf16 fragment-major)
//                 [+ bias[N]] [+ addend(MxN)] [relu]   (compile-time variants)
// Block = 128 threads (4 wave32). Block tile: 64 rows x 64 cols, full-K staged.
// grid.x = ceil(M/64), grid.y = NT/64.  KT in {32,64}, NT in {64,128,256}.
// Per wave: 1 B fragment (2x b128 from global, L2-resident) reused across
// 4 M-subtiles -> 8 static WMMAs for KT=64.
// ---------------------------------------------------------------------------
template <int KT, int NT, bool HAS_BIAS, bool HAS_ADD, bool RELU>
__global__ __launch_bounds__(128)
void gemm_bf16_wmma(const float* __restrict__ A, const __bf16* __restrict__ Bsw,
                    const float* __restrict__ bias, const float* __restrict__ addend,
                    float* __restrict__ C, int M) {
  __shared__ __align__(16) __bf16 AsB[64 * KT];  // A tile, bf16 (converted on stage)

  const int tid  = threadIdx.x;
  const int lane = tid & 31;
  const int wave = tid >> 5;
  const int m0   = blockIdx.x * 64;
  const int nt   = blockIdx.y * 4 + wave;   // global 16-column tile index
  const int r    = lane & 15;
  const int kh   = (lane < 16) ? 0 : 16;    // A-frag K base (contiguous k = 16h + j, matches B pack)

  // ---- stage A (64 x KT fp32 -> bf16), coalesced float4 loads, b64 LDS stores
  constexpr int ITER_A = (64 * KT) / (128 * 4);
#pragma unroll
  for (int c = 0; c < ITER_A; ++c) {
    int i   = (c * 128 + tid) * 4;
    int row = i / KT, col = i % KT;
    int rowc = m0 + row; if (rowc > M - 1) rowc = M - 1;   // clamp (no EXEC change)
    float4 v = *(const float4*)(A + (size_t)rowc * KT + col);
    uint2 p; p.x = pack2bf(v.x, v.y); p.y = pack2bf(v.z, v.w);
    *(uint2*)(&AsB[i]) = p;
  }
  __syncthreads();

  // ---- compute: 4 M-subtiles per wave, B fragment from global (frag-major) ----
  constexpr int KC = KT / 32;
  v8f acc[4] = {{}, {}, {}, {}};
#pragma unroll
  for (int kc = 0; kc < KC; ++kc) {
    bfrag32 braw = *(const bfrag32*)(Bsw + ((size_t)(nt * KC + kc) * 32 + lane) * 16);
    v16bf b = __builtin_bit_cast(v16bf, braw);
#pragma unroll
    for (int ms = 0; ms < 4; ++ms) {
      v16bf a;
#pragma unroll
      for (int j = 0; j < 16; ++j)          // two contiguous 16B runs -> 2x ds_load_b128
        a[j] = AsB[(ms * 16 + r) * KT + kc * 32 + kh + j];
      acc[ms] = __builtin_amdgcn_wmma_f32_16x16x32_bf16(false, a, false, b,
                                                        (short)0, acc[ms], false, false);
    }
  }

  // ---- epilogue: f32 C/D layout: VGPR v -> M = v + (lane<16 ? 0 : 8) ----
  const int n     = nt * 16 + r;
  const float bv  = HAS_BIAS ? bias[n] : 0.0f;
  const int moff  = (lane < 16) ? 0 : 8;
  float* Cp       = C + ((size_t)m0 + moff) * NT + n;
  const float* Ap = HAS_ADD ? (addend + ((size_t)m0 + moff) * NT + n) : nullptr;

  if (m0 + 64 <= M) {            // fast path: compile-time immediate offsets
#pragma unroll
    for (int ms = 0; ms < 4; ++ms) {
#pragma unroll
      for (int v = 0; v < 8; ++v) {
        float x = acc[ms][v] + bv;
        if (HAS_ADD) x += Ap[(size_t)((ms * 16 + v) * NT)];
        if (RELU)    x = fmaxf(x, 0.0f);
        Cp[(size_t)((ms * 16 + v) * NT)] = x;
      }
    }
  } else {
#pragma unroll
    for (int ms = 0; ms < 4; ++ms) {
#pragma unroll
      for (int v = 0; v < 8; ++v) {
        int m = m0 + moff + ms * 16 + v;
        if (m < M) {
          float x = acc[ms][v] + bv;
          if (HAS_ADD) x += Ap[(size_t)((ms * 16 + v) * NT)];
          if (RELU)    x = fmaxf(x, 0.0f);
          Cp[(size_t)((ms * 16 + v) * NT)] = x;
        }
      }
    }
  }
}

// ---------------------------------------------------------------------------
// Edge message + scatter-add: 64 threads per edge (one per feature).
// msg[f] = sum_b coeffs[etype,b] * hb[src, b*64+f];  atomicAdd(agg[tgt,f]).
// agg is 25.6 MB -> resident in the 192 MB L2; atomics land in L2.
// ---------------------------------------------------------------------------
__global__ __launch_bounds__(256)
void edge_scatter(const float* __restrict__ hb, const int* __restrict__ src,
                  const int* __restrict__ tgt, const int* __restrict__ etype,
                  const float* __restrict__ coeffs, float* __restrict__ agg, int E) {
  int gid = blockIdx.x * 256 + threadIdx.x;
  int e = gid >> 6;
  if (e >= E) return;
  int f = gid & 63;
  int s = src[e], t = tgt[e], rel = etype[e];
  const float* hrow = hb + (size_t)s * 256 + f;
  float c0 = coeffs[rel * 4 + 0], c1 = coeffs[rel * 4 + 1];
  float c2 = coeffs[rel * 4 + 2], c3 = coeffs[rel * 4 + 3];
  float m = c0 * hrow[0] + c1 * hrow[64] + c2 * hrow[128] + c3 * hrow[192];
  atomicAdd(agg + (size_t)t * 64 + f, m);
}

// ---------------------------------------------------------------------------
// Small prep / utility kernels
// ---------------------------------------------------------------------------
__global__ void k_zero(float* p, size_t n) {
  size_t i = (size_t)blockIdx.x * blockDim.x + threadIdx.x;
  if (i < n) p[i] = 0.0f;
}

__global__ void k_pad_rows(const float* __restrict__ in, float* __restrict__ out,
                           int M, int Kin, int Kout) {
  size_t i = (size_t)blockIdx.x * blockDim.x + threadIdx.x;
  if (i >= (size_t)M * Kout) return;
  int row = (int)(i / Kout), c = (int)(i % Kout);
  out[i] = (c < Kin) ? in[(size_t)row * Kin + c] : 0.0f;
}

// ---------------------------------------------------------------------------
// Pooling stage 1: per-block partial sum/max over 64 features.
// ---------------------------------------------------------------------------
__global__ __launch_bounds__(256)
void pool1(const float* __restrict__ h, float* __restrict__ red, int Nn) {
  __shared__ float ss[256], sm[256];
  int f = threadIdx.x & 63;
  int slot = threadIdx.x >> 6;
  float s = 0.0f, mx = -3.402823466e38f;
  for (int n = blockIdx.x * 4 + slot; n < Nn; n += gridDim.x * 4) {
    float v = h[(size_t)n * 64 + f];
    s += v; mx = fmaxf(mx, v);
  }
  ss[threadIdx.x] = s; sm[threadIdx.x] = mx;
  __syncthreads();
  if (slot == 0) {
    for (int k = 1; k < 4; ++k) { s += ss[k * 64 + f]; mx = fmaxf(mx, sm[k * 64 + f]); }
    red[blockIdx.x * 128 + f]      = s;
    red[blockIdx.x * 128 + 64 + f] = mx;
  }
}

// Pooling stage 2 + readout MLP (single block, 128 threads).
__global__ __launch_bounds__(128)
void pool2_mlp(const float* __restrict__ red, int nblk, int Nn,
               const float* __restrict__ Wr1, const float* __restrict__ br1,
               const float* __restrict__ Wr2, const float* __restrict__ br2,
               float* __restrict__ out) {
  __shared__ float g[128];
  __shared__ float t[64];
  int tid = threadIdx.x;
  if (tid < 64) {
    float s = 0.0f, mx = -3.402823466e38f;
    for (int p = 0; p < nblk; ++p) {
      s += red[p * 128 + tid];
      mx = fmaxf(mx, red[p * 128 + 64 + tid]);
    }
    g[tid]      = s / (float)Nn;   // mean pool
    g[64 + tid] = mx;              // max pool
  }
  __syncthreads();
  if (tid < 64) {
    float acc = br1[tid];
    for (int i = 0; i < 128; ++i) acc += g[i] * Wr1[(size_t)i * 64 + tid];
    t[tid] = fmaxf(acc, 0.0f);
  }
  __syncthreads();
  {
    float acc = br2[tid];
    for (int j = 0; j < 64; ++j) acc += t[j] * Wr2[(size_t)j * 128 + tid];
    out[tid] = acc;
  }
}

// ---------------------------------------------------------------------------
extern "C" void kernel_launch(void* const* d_in, const int* in_sizes, int n_in,
                              void* d_out, int out_size, void* d_ws, size_t ws_size,
                              hipStream_t stream) {
  (void)n_in; (void)out_size; (void)ws_size;

  const float* node_features = (const float*)d_in[0];
  const int*   edge_index    = (const int*)d_in[1];
  const int*   edge_type     = (const int*)d_in[2];
  const float* W_in   = (const float*)d_in[3];
  const float* b_in   = (const float*)d_in[4];
  const float* Wself0 = (const float*)d_in[5];
  const float* bself0 = (const float*)d_in[6];
  const float* bases0 = (const float*)d_in[7];
  const float* coeffs0= (const float*)d_in[8];
  const float* Wself1 = (const float*)d_in[9];
  const float* bself1 = (const float*)d_in[10];
  const float* bases1 = (const float*)d_in[11];
  const float* coeffs1= (const float*)d_in[12];
  const float* Wr1    = (const float*)d_in[13];
  const float* br1    = (const float*)d_in[14];
  const float* Wr2    = (const float*)d_in[15];
  const float* br2    = (const float*)d_in[16];
  const float* Wnp    = (const float*)d_in[17];
  const float* bnp    = (const float*)d_in[18];

  const int Nn = in_sizes[0] / 15;      // 100000
  const int E  = in_sizes[2];           // 1000000
  const int* src = edge_index;
  const int* tgt = edge_index + E;

  // ---- workspace layout (floats) ----
  float* ws = (float*)d_ws;
  size_t o = 0;
  float* Apad = ws + o; o += (size_t)Nn * 32;    // padded features
  float* h    = ws + o; o += (size_t)Nn * 64;    // activations (ping)
  float* h2   = ws + o; o += (size_t)Nn * 64;    // activations (pong)
  float* hb   = ws + o; o += (size_t)Nn * 256;   // per-basis projections
  float* agg  = ws + o; o += (size_t)Nn * 64;    // scatter-add target
  float* red  = ws + o; o += (size_t)512 * 128;  // pooling partials
  __bf16* wb   = (__bf16*)(ws + o);              // fragment-major bf16 weights
  __bf16* Winb = wb;                 // 32*64   = 2048
  __bf16* Bc0  = Winb + 2048;        // 64*256  = 16384
  __bf16* Bc1  = Bc0 + 16384;
  __bf16* Ws0b = Bc1 + 16384;        // 64*64   = 4096
  __bf16* Ws1b = Ws0b + 4096;
  __bf16* Wnpb = Ws1b + 4096;        // 64*128  = 8192

  // ---- weight prep: pack into fragment-major bf16 ----
  k_pad_rows<<<(unsigned)(((size_t)Nn * 32 + 255) / 256), 256, 0, stream>>>(
      node_features, Apad, Nn, 15, 32);
  k_pack_B<32,  64><<< 8, 256, 0, stream>>>(W_in,   Winb, 15, 0);
  k_pack_B<64, 256><<<64, 256, 0, stream>>>(bases0, Bc0,  64, 1);
  k_pack_B<64, 256><<<64, 256, 0, stream>>>(bases1, Bc1,  64, 1);
  k_pack_B<64,  64><<<16, 256, 0, stream>>>(Wself0, Ws0b, 64, 0);
  k_pack_B<64,  64><<<16, 256, 0, stream>>>(Wself1, Ws1b, 64, 0);
  k_pack_B<64, 128><<<32, 256, 0, stream>>>(Wnp,    Wnpb, 64, 0);

  const unsigned MT = (unsigned)((Nn + 63) / 64);

  // ---- input projection: h = Apad @ W_in + b_in ----
  gemm_bf16_wmma<32, 64, true, false, false><<<dim3(MT, 1), 128, 0, stream>>>(
      Apad, Winb, b_in, nullptr, h, Nn);

  // ---- two RGCN layers ----
  const __bf16* BcL[2]  = {Bc0, Bc1};
  const __bf16* WsLb[2] = {Ws0b, Ws1b};
  const float*  bsL[2]  = {bself0, bself1};
  const float*  cfL[2]  = {coeffs0, coeffs1};
  float* hin = h; float* hout = h2;
  for (int L = 0; L < 2; ++L) {
    k_zero<<<(unsigned)(((size_t)Nn * 64 + 255) / 256), 256, 0, stream>>>(
        agg, (size_t)Nn * 64);
    gemm_bf16_wmma<64, 256, false, false, false><<<dim3(MT, 4), 128, 0, stream>>>(
        hin, BcL[L], nullptr, nullptr, hb, Nn);
    edge_scatter<<<(unsigned)(((size_t)E * 64 + 255) / 256), 256, 0, stream>>>(
        hb, src, tgt, edge_type, cfL[L], agg, E);
    gemm_bf16_wmma<64, 64, true, true, true><<<dim3(MT, 1), 128, 0, stream>>>(
        hin, WsLb[L], bsL[L], agg, hout, Nn);
    float* tmp = hin; hin = hout; hout = tmp;
  }

  // ---- readout: pooling + MLP -> graph_emb (d_out[0:128]) ----
  pool1<<<512, 256, 0, stream>>>(hin, red, Nn);
  pool2_mlp<<<1, 128, 0, stream>>>(red, 512, Nn, Wr1, br1, Wr2, br2, (float*)d_out);

  // ---- node_emb = h @ Wnp + bnp -> d_out[128 : 128 + N*128] ----
  gemm_bf16_wmma<64, 128, true, false, false><<<dim3(MT, 2), 128, 0, stream>>>(
      hin, Wnpb, bnp, nullptr, (float*)d_out + 128, Nn);
}
